// AttentionBlock_46213848105805
// MI455X (gfx1250) — matrix operand order repaired and checked by reading, not compiled, not run on hardware
//
#include <hip/hip_runtime.h>
#include <stdint.h>

// ---------- types ----------
typedef __attribute__((ext_vector_type(16))) __bf16    v16bf;
typedef __attribute__((ext_vector_type(8)))  float     v8f;
typedef __attribute__((ext_vector_type(4)))  unsigned  u32x4;
typedef __attribute__((ext_vector_type(8)))  unsigned  u32x8;

// Async global->LDS path (ASYNCcnt) if the toolchain exposes the builtins.
#ifdef __has_builtin
#if __has_builtin(__builtin_amdgcn_global_load_async_to_lds_b128) && \
    __has_builtin(__builtin_amdgcn_s_wait_asynccnt)
#define USE_ASYNC_LDS 1
#endif
#endif

#ifdef USE_ASYNC_LDS
// Builtin prototype (from clang diagnostics):
//   void __builtin_amdgcn_global_load_async_to_lds_b128(
//       int __attribute__((vector_size(16))) addrspace(1)* src,
//       int __attribute__((vector_size(16))) addrspace(3)* dst,
//       imm int offset, imm int cpol);
typedef int async_v4i __attribute__((vector_size(16)));
#define GLD_ASYNC_B128(ldsdst, gsrc)                                          \
    __builtin_amdgcn_global_load_async_to_lds_b128(                           \
        (__attribute__((address_space(1))) async_v4i*)(gsrc),                 \
        (__attribute__((address_space(3))) async_v4i*)(ldsdst), 0, 0)
#endif

struct FragSplit { u32x4 lo, hi; };   // 32 bytes, two 16B chunks

static __device__ __forceinline__ unsigned short f2bf(float f) {
    unsigned u = __builtin_bit_cast(unsigned, f);
    u += 0x7fffu + ((u >> 16) & 1u);              // round-to-nearest-even
    return (unsigned short)(u >> 16);
}

// A-fragment (split-K-chunk) load: two contiguous 16B chunks.
static __device__ __forceinline__ v16bf ld_frag_split(const unsigned short* p0,
                                                      const unsigned short* p1) {
    FragSplit f;
    f.lo = *(const u32x4*)p0;
    f.hi = *(const u32x4*)p1;
    return __builtin_bit_cast(v16bf, f);
}

// B-fragment load: one contiguous 32B chunk.
static __device__ __forceinline__ v16bf ld_frag_contig(const unsigned short* p) {
    u32x8 x = *(const u32x8*)p;
    return __builtin_bit_cast(v16bf, x);
}

static __device__ __forceinline__ v8f wmma_bf16(v16bf a, v16bf b, v8f c) {
    return __builtin_amdgcn_wmma_f32_16x16x32_bf16(false, a, false, b,
                                                   (short)0, c, false, false);
}

// ---------- problem constants ----------
#define BATCH 2
#define SEQ   2048
#define EMB   1024
#define HEADS 16
#define DH    64
#define ROWS  (BATCH * SEQ)     // 4096
#define SCALE 8.0f              // sqrt(DH), reproduced faithfully
#define LN_EPS 1e-5f

// ---------------------------------------------------------------------------
// Kernel 1: LayerNorm + bf16 cast.  One block per row of 1024.
// ---------------------------------------------------------------------------
__global__ __launch_bounds__(256) void ln_cast_kernel(
    const float* __restrict__ x, const float* __restrict__ g,
    const float* __restrict__ bb, unsigned short* __restrict__ hbf) {
    __shared__ float rs[256], rs2[256];
    const int row = blockIdx.x, tid = threadIdx.x;
    const float* xr = x + (size_t)row * EMB;
    float4 v = ((const float4*)xr)[tid];
    float s  = v.x + v.y + v.z + v.w;
    float s2 = v.x * v.x + v.y * v.y + v.z * v.z + v.w * v.w;
    rs[tid] = s; rs2[tid] = s2;
    __syncthreads();
    for (int off = 128; off > 0; off >>= 1) {
        if (tid < off) { rs[tid] += rs[tid + off]; rs2[tid] += rs2[tid + off]; }
        __syncthreads();
    }
    const float mean = rs[0] * (1.0f / EMB);
    const float var  = rs2[0] * (1.0f / EMB) - mean * mean;
    const float rstd = rsqrtf(var + LN_EPS);
    const int c = tid * 4;
    float h0 = (v.x - mean) * rstd * g[c + 0] + bb[c + 0];
    float h1 = (v.y - mean) * rstd * g[c + 1] + bb[c + 1];
    float h2 = (v.z - mean) * rstd * g[c + 2] + bb[c + 2];
    float h3 = (v.w - mean) * rstd * g[c + 3] + bb[c + 3];
    uint2 pk;
    pk.x = (unsigned)f2bf(h0) | ((unsigned)f2bf(h1) << 16);
    pk.y = (unsigned)f2bf(h2) | ((unsigned)f2bf(h3) << 16);
    ((uint2*)(hbf + (size_t)row * EMB))[tid] = pk;
}

// ---------------------------------------------------------------------------
// Kernel 2: cast the 4 weight matrices (fp32 -> bf16), slice per blockIdx.y.
// ---------------------------------------------------------------------------
__global__ __launch_bounds__(256) void wcast_kernel(
    const float* __restrict__ wq, const float* __restrict__ wk,
    const float* __restrict__ wv, const float* __restrict__ wo,
    unsigned short* __restrict__ out) {
    const float* W = (blockIdx.y == 0) ? wq : (blockIdx.y == 1) ? wk
                   : (blockIdx.y == 2) ? wv : wo;
    const size_t base = (size_t)blockIdx.y * (EMB * EMB);
    const int idx = (blockIdx.x * 256 + threadIdx.x) * 4;
    float4 v = *(const float4*)(W + idx);
    uint2 pk;
    pk.x = (unsigned)f2bf(v.x) | ((unsigned)f2bf(v.y) << 16);
    pk.y = (unsigned)f2bf(v.z) | ((unsigned)f2bf(v.w) << 16);
    *(uint2*)(out + base + idx) = pk;
}

// ---------------------------------------------------------------------------
// Kernel 3: QKV projections.  y = h @ W^T + b.
// Register-blocked: one wave computes a 32x64 output tile
// (2 A-frags x 4 B-frags -> 8 WMMA per 12 b128 loads).
// z selects Q/K/V.  Q,K stored [B,H,L,DH]; V stored transposed [B,H,DH,L].
// ---------------------------------------------------------------------------
__global__ __launch_bounds__(256) void qkv_gemm_kernel(
    const unsigned short* __restrict__ hbf, const unsigned short* __restrict__ wbf,
    const float* __restrict__ bq, const float* __restrict__ bk,
    const float* __restrict__ bv,
    unsigned short* __restrict__ qb, unsigned short* __restrict__ kbf,
    unsigned short* __restrict__ vtb) {
    const int wave = threadIdx.x >> 5, lane = threadIdx.x & 31;
    const int hf = lane >> 4, l16 = lane & 15;
    const int mt  = blockIdx.x;                 // 32-row tile   (0..127)
    const int n64 = blockIdx.y * 8 + wave;      // 64-col tile   (0..15)
    const int z   = blockIdx.z;
    const unsigned short* W = wbf + (size_t)z * (EMB * EMB);
    const float* bias = (z == 0) ? bq : (z == 1) ? bk : bv;

    const unsigned short* arow0 = hbf + (size_t)(mt * 32 + l16) * EMB + hf * 16;
    const unsigned short* arow1 = arow0 + (size_t)16 * EMB;
    const unsigned short* brow0 = W + (size_t)(n64 * 64 + l16) * EMB + hf * 16;

    v8f acc[2][4] = {};
    for (int k0 = 0; k0 < EMB; k0 += 32) {
        v16bf a0 = ld_frag_split(arow0 + k0, arow0 + k0 + 8);
        v16bf a1 = ld_frag_split(arow1 + k0, arow1 + k0 + 8);
#pragma unroll
        for (int n = 0; n < 4; ++n) {
            v16bf b = ld_frag_contig(brow0 + (size_t)n * 16 * EMB + k0);
            acc[0][n] = wmma_bf16(a0, b, acc[0][n]);
            acc[1][n] = wmma_bf16(a1, b, acc[1][n]);
        }
    }

#pragma unroll
    for (int n = 0; n < 4; ++n) {
        const int col = n64 * 64 + n * 16 + l16;  // column of E
        const int hh  = col >> 6;                 // head
        const int d   = col & 63;                 // within-head dim
        const float bcol = bias[col];
#pragma unroll
        for (int mh = 0; mh < 2; ++mh) {
#pragma unroll
            for (int j = 0; j < 8; ++j) {
                const int m  = mt * 32 + mh * 16 + j + 8 * hf;  // row (b*L + l)
                const int b_ = m >> 11, l = m & (SEQ - 1);
                const unsigned short bf = f2bf(acc[mh][n][j] + bcol);
                if (z == 0)
                    qb [((size_t)(b_ * HEADS + hh) * SEQ + l) * DH + d] = bf;
                else if (z == 1)
                    kbf[((size_t)(b_ * HEADS + hh) * SEQ + l) * DH + d] = bf;
                else
                    vtb[((size_t)(b_ * HEADS + hh) * DH + d) * SEQ + l] = bf;
            }
        }
    }
}

// ---------------------------------------------------------------------------
// Kernel 4: flash attention.  8 waves per block share one (b,h); each wave
// owns a 16-query tile.  K/V 32-key tiles are staged cooperatively into
// double-buffered LDS (one b128 per thread per tile).  When available, the
// staging uses GLOBAL_LOAD_ASYNC_TO_LDS_B128 + s_wait_asynccnt so the copy
// bypasses VGPRs and overlaps with the WMMA stream; otherwise it falls back
// to a register double-buffer.  Online softmax; P restriped via 1KB LDS/wave.
// ---------------------------------------------------------------------------
__global__ __launch_bounds__(256) void attn_kernel(
    const unsigned short* __restrict__ qb, const unsigned short* __restrict__ kbf,
    const unsigned short* __restrict__ vtb, unsigned short* __restrict__ ob) {
    __shared__ __align__(16) unsigned short kT[2][32 * DH];   // keys x d
    __shared__ __align__(16) unsigned short vT[2][DH * 32];   // d x keys
    __shared__ __align__(16) unsigned short P[8][16 * 32];
    const int tid = threadIdx.x;
    const int wave = tid >> 5, lane = tid & 31;
    const int hf = lane >> 4, l16 = lane & 15;
    const int w  = blockIdx.x * 8 + wave;
    const int b  = w >> 11;              // 2048 waves per batch
    const int h  = (w >> 7) & 15;        // 128 q-tiles per head
    const int qt = w & 127;
    const size_t bh = (size_t)(b * HEADS + h);   // uniform across the block

    // Q A-fragments (d = 0..31 and 32..63), resident for the whole loop.
    const unsigned short* qbase = qb + (bh * SEQ + qt * 16 + l16) * DH + hf * 16;
    const v16bf qa0 = ld_frag_split(qbase,      qbase + 8);
    const v16bf qa1 = ld_frag_split(qbase + 32, qbase + 40);

    const unsigned short* kbase = kbf + bh * (size_t)SEQ * DH;
    const unsigned short* vbase = vtb + bh * (size_t)DH * SEQ;

    // cooperative staging roles: 16B per thread per tile
    const int krow = tid >> 3, kchunk = tid & 7;   // K: 32 rows x 128B
    const int vrow = tid >> 2, vchunk = tid & 3;   // V: 64 rows x 64B
    const unsigned short* kg = kbase + (size_t)krow * DH + kchunk * 8;
    const unsigned short* vg = vbase + (size_t)vrow * SEQ + vchunk * 8;
    unsigned short* kl = &kT[0][krow * DH + kchunk * 8];
    unsigned short* vl = &vT[0][vrow * 32 + vchunk * 8];
    const int kbufstride = 32 * DH, vbufstride = DH * 32;

    unsigned short* pl = &P[wave][0];

    float mi[8], li[8];
    v8f o0 = {}, o1 = {}, o2 = {}, o3 = {};
#pragma unroll
    for (int j = 0; j < 8; ++j) { mi[j] = -3.0e38f; li[j] = 0.0f; }

    // prologue: stage tile 0 into buffer 0
#ifdef USE_ASYNC_LDS
    GLD_ASYNC_B128(kl, kg);
    GLD_ASYNC_B128(vl, vg);
#else
    u32x4 kr = *(const u32x4*)kg;
    u32x4 vr = *(const u32x4*)vg;
    *(u32x4*)kl = kr;
    *(u32x4*)vl = vr;
#endif

    for (int it = 0; it < SEQ / 32; ++it) {
        const int k0 = it * 32;
        const bool more = (it + 1) < (SEQ / 32);
        const int nb = (it + 1) & 1;
#ifdef USE_ASYNC_LDS
        __builtin_amdgcn_s_wait_asynccnt(0);  // our async copies have landed
        __syncthreads();                      // everyone's copies + readers done
        if (more) {                           // kick off next tile's DMA
            GLD_ASYNC_B128(kl + nb * kbufstride, kg + (size_t)(k0 + 32) * DH);
            GLD_ASYNC_B128(vl + nb * vbufstride, vg + (k0 + 32));
        }
#else
        if (more) {   // fetch next tile into registers while LDS settles
            kr = *(const u32x4*)(kg + (size_t)(k0 + 32) * DH);
            vr = *(const u32x4*)(vg + (k0 + 32));
        }
        __syncthreads();   // buf[it&1] stores visible; buf[nb] readers done
        if (more) {
            *(u32x4*)(kl + nb * kbufstride) = kr;
            *(u32x4*)(vl + nb * vbufstride) = vr;
        }
#endif
        const unsigned short* kc = kT[it & 1];
        const unsigned short* vc = vT[it & 1];

        // scores: S = Q(16x64) @ K_tile^T(64x16), two 16-key tiles
        v8f s0 = {}, s1 = {};
        s0 = wmma_bf16(qa0, ld_frag_contig(&kc[(l16)      * DH      + hf * 16]), s0);
        s0 = wmma_bf16(qa1, ld_frag_contig(&kc[(l16)      * DH + 32 + hf * 16]), s0);
        s1 = wmma_bf16(qa0, ld_frag_contig(&kc[(16 + l16) * DH      + hf * 16]), s1);
        s1 = wmma_bf16(qa1, ld_frag_contig(&kc[(16 + l16) * DH + 32 + hf * 16]), s1);

        float alpha[8];
#pragma unroll
        for (int j = 0; j < 8; ++j) {
            float a0 = s0[j] * SCALE;
            float a1 = s1[j] * SCALE;
            // row max across the 16 lanes of this half
            float r = fmaxf(a0, a1);
            r = fmaxf(r, __shfl_xor(r, 1, 32));
            r = fmaxf(r, __shfl_xor(r, 2, 32));
            r = fmaxf(r, __shfl_xor(r, 4, 32));
            r = fmaxf(r, __shfl_xor(r, 8, 32));
            const float mn = fmaxf(mi[j], r);
            alpha[j] = __expf(mi[j] - mn);
            mi[j] = mn;
            const float p0 = __expf(a0 - mn);
            const float p1 = __expf(a1 - mn);
            float rsum = p0 + p1;
            rsum += __shfl_xor(rsum, 1, 32);
            rsum += __shfl_xor(rsum, 2, 32);
            rsum += __shfl_xor(rsum, 4, 32);
            rsum += __shfl_xor(rsum, 8, 32);
            li[j] = li[j] * alpha[j] + rsum;
            // restripe P (C/D layout -> row-major 16x32 in LDS)
            const int m = j + 8 * hf;
            pl[m * 32 + l16]      = f2bf(p0);
            pl[m * 32 + 16 + l16] = f2bf(p1);
        }

        // P A-fragment (16x32) from LDS
        const unsigned short* pb = pl + l16 * 32 + hf * 16;
        const v16bf pa = ld_frag_split(pb, pb + 8);

        // rescale accumulators, then O += P @ V (4 column tiles of 16)
#pragma unroll
        for (int j = 0; j < 8; ++j) {
            o0[j] *= alpha[j]; o1[j] *= alpha[j];
            o2[j] *= alpha[j]; o3[j] *= alpha[j];
        }
        o0 = wmma_bf16(pa, ld_frag_contig(&vc[(0 * 16 + l16) * 32 + hf * 16]), o0);
        o1 = wmma_bf16(pa, ld_frag_contig(&vc[(1 * 16 + l16) * 32 + hf * 16]), o1);
        o2 = wmma_bf16(pa, ld_frag_contig(&vc[(2 * 16 + l16) * 32 + hf * 16]), o2);
        o3 = wmma_bf16(pa, ld_frag_contig(&vc[(3 * 16 + l16) * 32 + hf * 16]), o3);
    }

    // epilogue: normalize and store O as [B, L, E] (head-major within E)
#pragma unroll
    for (int j = 0; j < 8; ++j) {
        const float inv = 1.0f / li[j];
        const int m = j + 8 * hf;
        const int l = qt * 16 + m;
        unsigned short* orow = ob + ((size_t)b * SEQ + l) * EMB + h * DH + l16;
        orow[0]  = f2bf(o0[j] * inv);
        orow[16] = f2bf(o1[j] * inv);
        orow[32] = f2bf(o2[j] * inv);
        orow[48] = f2bf(o3[j] * inv);
    }
}

// ---------------------------------------------------------------------------
// Kernel 5: output projection + bias + residual (fp32 out), 32x64 per wave.
// ---------------------------------------------------------------------------
__global__ __launch_bounds__(256) void out_gemm_kernel(
    const unsigned short* __restrict__ ob, const unsigned short* __restrict__ wo,
    const float* __restrict__ bo, const float* __restrict__ x,
    float* __restrict__ out) {
    const int wave = threadIdx.x >> 5, lane = threadIdx.x & 31;
    const int hf = lane >> 4, l16 = lane & 15;
    const int mt  = blockIdx.x;
    const int n64 = blockIdx.y * 8 + wave;

    const unsigned short* arow0 = ob + (size_t)(mt * 32 + l16) * EMB + hf * 16;
    const unsigned short* arow1 = arow0 + (size_t)16 * EMB;
    const unsigned short* brow0 = wo + (size_t)(n64 * 64 + l16) * EMB + hf * 16;

    v8f acc[2][4] = {};
    for (int k0 = 0; k0 < EMB; k0 += 32) {
        v16bf a0 = ld_frag_split(arow0 + k0, arow0 + k0 + 8);
        v16bf a1 = ld_frag_split(arow1 + k0, arow1 + k0 + 8);
#pragma unroll
        for (int n = 0; n < 4; ++n) {
            v16bf b = ld_frag_contig(brow0 + (size_t)n * 16 * EMB + k0);
            acc[0][n] = wmma_bf16(a0, b, acc[0][n]);
            acc[1][n] = wmma_bf16(a1, b, acc[1][n]);
        }
    }

#pragma unroll
    for (int n = 0; n < 4; ++n) {
        const int col = n64 * 64 + n * 16 + l16;
        const float bcol = bo[col];
#pragma unroll
        for (int mh = 0; mh < 2; ++mh) {
#pragma unroll
            for (int j = 0; j < 8; ++j) {
                const int m = mt * 32 + mh * 16 + j + 8 * hf;
                const size_t idx = (size_t)m * EMB + col;
                out[idx] = acc[mh][n][j] + bcol + x[idx];
            }
        }
    }
}

// ---------------------------------------------------------------------------
extern "C" void kernel_launch(void* const* d_in, const int* in_sizes, int n_in,
                              void* d_out, int out_size, void* d_ws, size_t ws_size,
                              hipStream_t stream) {
    (void)in_sizes; (void)n_in; (void)out_size; (void)ws_size;
    const float* x    = (const float*)d_in[0];
    const float* ln_g = (const float*)d_in[1];
    const float* ln_b = (const float*)d_in[2];
    const float* Wq   = (const float*)d_in[3];
    const float* bq   = (const float*)d_in[4];
    const float* Wk   = (const float*)d_in[5];
    const float* bk   = (const float*)d_in[6];
    const float* Wv   = (const float*)d_in[7];
    const float* bv   = (const float*)d_in[8];
    const float* Wo   = (const float*)d_in[9];
    const float* bo   = (const float*)d_in[10];
    float* out = (float*)d_out;

    const size_t MB = 1024 * 1024;
    char* ws = (char*)d_ws;
    unsigned short* hbf = (unsigned short*)(ws);             //  8 MiB: LN(x) bf16
    unsigned short* wbf = (unsigned short*)(ws + 8 * MB);    //  8 MiB: Wq,Wk,Wv,Wo bf16
    unsigned short* qbv = (unsigned short*)(ws + 16 * MB);   //  8 MiB: Q [B,H,L,DH]
    unsigned short* kbf = (unsigned short*)(ws + 24 * MB);   //  8 MiB: K [B,H,L,DH]
    unsigned short* vtb = (unsigned short*)(ws + 32 * MB);   //  8 MiB: V [B,H,DH,L]
    unsigned short* obf = (unsigned short*)(ws + 40 * MB);   //  8 MiB: O [B,L,E]

    ln_cast_kernel<<<ROWS, 256, 0, stream>>>(x, ln_g, ln_b, hbf);
    wcast_kernel<<<dim3(EMB * EMB / (256 * 4), 4), 256, 0, stream>>>(Wq, Wk, Wv, Wo, wbf);
    qkv_gemm_kernel<<<dim3(ROWS / 32, 2, 3), 256, 0, stream>>>(
        hbf, wbf, bq, bk, bv, qbv, kbf, vtb);
    attn_kernel<<<dim3(BATCH * HEADS * (SEQ / 16) / 8), 256, 0, stream>>>(
        qbv, kbf, vtb, obf);
    out_gemm_kernel<<<dim3(ROWS / 32, 2), 256, 0, stream>>>(
        obf, wbf + 3 * (size_t)(EMB * EMB), bo, x, out);
}
